// MoELayer_3779571220990
// MI455X (gfx1250) — matrix operand order repaired and checked by reading, not compiled, not run on hardware
//
#include <hip/hip_runtime.h>
#include <hip/hip_bf16.h>

// ---------------------------------------------------------------------------
// MoE forward for MI455X (gfx1250, wave32, WMMA bf16 w/ f32 accumulate)
//   D_MODEL=1024, D_FF=4096, E=8, TOP_K=2, T=8192
// ---------------------------------------------------------------------------

#define DM 1024
#define DF 4096
#define NE 8
#define NT 8192   // tokens
#define ROWS 32   // tokens per expert tile

typedef __attribute__((ext_vector_type(16))) __bf16 v16bf;
typedef __attribute__((ext_vector_type(8)))  float  v8f;
typedef __attribute__((ext_vector_type(4)))  int    v4i;
typedef __attribute__((address_space(1))) v4i* gv4i_p;   // global
typedef __attribute__((address_space(3))) v4i* lv4i_p;   // LDS

union AFrag { v16bf v; uint4 q[2]; };

__device__ __forceinline__ unsigned short f2bf(float f) {
  unsigned u = __float_as_uint(f);
  unsigned r = 0x7FFFu + ((u >> 16) & 1u);   // round-to-nearest-even
  return (unsigned short)((u + r) >> 16);
}

// --- CDNA5 async global->LDS copy (ASYNCcnt path) --------------------------
__device__ __forceinline__ void async_copy16(const unsigned short* g,
                                             unsigned short* l) {
#if __has_builtin(__builtin_amdgcn_global_load_async_to_lds_b128)
  __builtin_amdgcn_global_load_async_to_lds_b128(
      (gv4i_p)(v4i*)g, (lv4i_p)(v4i*)l, 0, 0);
#else
  asm volatile("global_load_async_to_lds_b128 %0, %1, off"
               :: "v"((unsigned)(unsigned long long)l),
                  "v"((unsigned long long)g)
               : "memory");
#endif
}

__device__ __forceinline__ void async_wait0() {
#if __has_builtin(__builtin_amdgcn_s_wait_asynccnt)
  __builtin_amdgcn_s_wait_asynccnt(0);
#else
  asm volatile("s_wait_asynccnt 0x0" ::: "memory");
#endif
}

// ---------------------------------------------------------------------------
// 0) f32 -> bf16 flat conversion (for x)
// ---------------------------------------------------------------------------
__global__ void convert_bf16(const float* __restrict__ src,
                             unsigned short* __restrict__ dst) {
  size_t i = ((size_t)blockIdx.x * 256 + threadIdx.x) * 8;
  const float4* s = (const float4*)(src + i);
  float4 v0 = s[0], v1 = s[1];
  uint4 pk;
  pk.x = (unsigned)f2bf(v0.x) | ((unsigned)f2bf(v0.y) << 16);
  pk.y = (unsigned)f2bf(v0.z) | ((unsigned)f2bf(v0.w) << 16);
  pk.z = (unsigned)f2bf(v1.x) | ((unsigned)f2bf(v1.y) << 16);
  pk.w = (unsigned)f2bf(v1.z) | ((unsigned)f2bf(v1.w) << 16);
  *(uint4*)(dst + i) = pk;
}

// ---------------------------------------------------------------------------
// 1) f32 -> bf16 transposed weight panels: src[R][C] f32 -> dst[C][R] bf16
// ---------------------------------------------------------------------------
__global__ void transpose_bf16(const float* __restrict__ src,
                               unsigned short* __restrict__ dst,
                               int R, int C) {
  __shared__ float tile[32][33];
  const float* s = src + (size_t)blockIdx.z * R * C;
  unsigned short* d = dst + (size_t)blockIdx.z * R * C;
  int c0 = blockIdx.x * 32, r0 = blockIdx.y * 32;
#pragma unroll
  for (int j = 0; j < 32; j += 8) {
    int r = r0 + threadIdx.y + j;
    tile[threadIdx.y + j][threadIdx.x] = s[(size_t)r * C + c0 + threadIdx.x];
  }
  __syncthreads();
#pragma unroll
  for (int j = 0; j < 32; j += 8) {
    int c = c0 + threadIdx.y + j;
    d[(size_t)c * R + r0 + threadIdx.x] = f2bf(tile[threadIdx.x][threadIdx.y + j]);
  }
}

// ---------------------------------------------------------------------------
// 2) Router: one wave32 per token -> combine[T][8]
// ---------------------------------------------------------------------------
__global__ void router_kernel(const float* __restrict__ x,
                              const float* __restrict__ Wg,
                              const float* __restrict__ bg,
                              float* __restrict__ combine) {
  int wave = threadIdx.x >> 5;
  int lane = threadIdx.x & 31;
  int t = blockIdx.x * 8 + wave;
  const float* xr = x + (size_t)t * DM;
  float acc[NE] = {};
  for (int it = 0; it < DM / 32; ++it) {
    int dpos = it * 32 + lane;
    float xv = xr[dpos];
    const float* wg = Wg + dpos * NE;
#pragma unroll
    for (int e = 0; e < NE; ++e) acc[e] += xv * wg[e];
  }
#pragma unroll
  for (int e = 0; e < NE; ++e)
    for (int off = 16; off > 0; off >>= 1)
      acc[e] += __shfl_xor(acc[e], off, 32);
  if (lane == 0) {
    float l[NE], m = -1e30f;
#pragma unroll
    for (int e = 0; e < NE; ++e) { l[e] = acc[e] + bg[e]; m = fmaxf(m, l[e]); }
    float p[NE];
#pragma unroll
    for (int e = 0; e < NE; ++e) p[e] = __expf(l[e] - m);
    int i1 = 0;
#pragma unroll
    for (int e = 1; e < NE; ++e) if (p[e] > p[i1]) i1 = e;
    int i2 = (i1 == 0) ? 1 : 0;
#pragma unroll
    for (int e = 0; e < NE; ++e) if (e != i1 && p[e] > p[i2]) i2 = e;
    float ws = p[i1] + p[i2];
    float* c = combine + (size_t)t * NE;
#pragma unroll
    for (int e = 0; e < NE; ++e) c[e] = 0.f;
    c[i1] = p[i1] / ws;
    c[i2] = p[i2] / ws;
  }
}

// ---------------------------------------------------------------------------
// 3) Build per-expert token lists
// ---------------------------------------------------------------------------
__global__ void assign_kernel(const float* __restrict__ combine,
                              int* __restrict__ counts,
                              int* __restrict__ tokens) {
  int t = blockIdx.x * 256 + threadIdx.x;
#pragma unroll
  for (int e = 0; e < NE; ++e) {
    float w = combine[(size_t)t * NE + e];
    if (w > 0.f) {
      int pos = atomicAdd(&counts[e], 1);
      tokens[e * NT + pos] = t;
    }
  }
}

// ---------------------------------------------------------------------------
// 4) Fused two-layer expert GEMM. LDS: X tile 64KB + H tile 256KB = 320KB.
//    16B-chunk XOR swizzle kills bank conflicts on row-strided A-frag loads.
// ---------------------------------------------------------------------------
__global__ __launch_bounds__(256, 1)
void moe_expert_kernel(const unsigned short* __restrict__ xb,  // [NT][DM] bf16
                       const unsigned short* __restrict__ W1t, // [E][DF][DM] bf16 (N,K)
                       const float* __restrict__ b1,
                       const unsigned short* __restrict__ W2t, // [E][DM][DF] bf16 (N,K)
                       const float* __restrict__ b2,
                       const float* __restrict__ combine,
                       const int* __restrict__ counts,
                       const int* __restrict__ tokens,
                       float* __restrict__ out) {
  extern __shared__ unsigned short smem[];
  unsigned short* Xs = smem;              // [32][1024] bf16, swizzled
  unsigned short* Hs = smem + ROWS * DM;  // [32][4096] bf16, swizzled

  const int e = blockIdx.y;
  const int cnt = counts[e];
  const int base = blockIdx.x * ROWS;
  if (base >= cnt) return;
  const int* tlist = tokens + e * NT + base;
  const int tid = threadIdx.x;

  // ---- Phase 0: async gather of bf16 token rows -> swizzled LDS ----
  for (int i = tid; i < ROWS * (DM / 8); i += 256) {
    int r = i >> 7;                 // row 0..31
    int c = i & 127;                // 16B chunk within row (8 bf16)
    int rr = (base + r < cnt) ? r : 0;
    int tok = tlist[rr];
    async_copy16(xb + (size_t)tok * DM + c * 8,
                 Xs + r * DM + ((c ^ (r & 15)) * 8));
  }
  async_wait0();
  __syncthreads();

  const int wave = tid >> 5, lane = tid & 31;
  const int lh = (lane >> 4) & 1;   // lane half
  const int ln = lane & 15;
  // A-matrix 16x32 bf16 split-K layout (ISA 7.12.2)
  const int akoff0 = lh ? 8 : 0;
  const int akoff1 = lh ? 24 : 16;
  // B-matrix 32x16: same split-K layout as A
  const int bkoff = lh ? 8 : 0;

  // ---- Phase 1: H = relu(X @ W1[e] + b1[e]), wave owns 512 of 4096 cols ----
  for (int nt = 0; nt < 32; ++nt) {
    int n = wave * 512 + nt * 16 + ln;
    float bias = b1[e * DF + n];
    v8f acc0, acc1;
#pragma unroll
    for (int r = 0; r < 8; ++r) { acc0[r] = bias; acc1[r] = bias; }
    const unsigned short* wrow = W1t + ((size_t)e * DF + n) * DM;
    if (nt < 31) __builtin_prefetch(wrow + 16 * DM, 0, 1);  // next N-tile row
#pragma unroll 4
    for (int k = 0; k < DM; k += 32) {
      AFrag bf;
      bf.q[0] = *(const uint4*)(wrow + k + bkoff);
      bf.q[1] = *(const uint4*)(wrow + k + bkoff + 16);
      AFrag a0, a1;
      int c0 = (k + akoff0) >> 3, c1 = (k + akoff1) >> 3;
      a0.q[0] = *(const uint4*)(Xs + ln * DM + ((c0 ^ ln) * 8));
      a0.q[1] = *(const uint4*)(Xs + ln * DM + ((c1 ^ ln) * 8));
      a1.q[0] = *(const uint4*)(Xs + (16 + ln) * DM + ((c0 ^ ln) * 8));
      a1.q[1] = *(const uint4*)(Xs + (16 + ln) * DM + ((c1 ^ ln) * 8));
      acc0 = __builtin_amdgcn_wmma_f32_16x16x32_bf16(false, a0.v, false, bf.v,
                                                     (short)0, acc0, false, false);
      acc1 = __builtin_amdgcn_wmma_f32_16x16x32_bf16(false, a1.v, false, bf.v,
                                                     (short)0, acc1, false, false);
    }
    // relu -> bf16 -> Hs (C-layout: lane=N, vgpr r -> M = r + lh*8)
#pragma unroll
    for (int r = 0; r < 8; ++r) {
      int m0 = r + lh * 8;
      Hs[m0 * DF + (((n >> 3) ^ (m0 & 15)) * 8) + (n & 7)] = f2bf(fmaxf(acc0[r], 0.f));
      int m1 = m0 + 16;
      Hs[m1 * DF + (((n >> 3) ^ (m1 & 15)) * 8) + (n & 7)] = f2bf(fmaxf(acc1[r], 0.f));
    }
  }
  __syncthreads();

  // ---- Phase 2: Y = H @ W2[e] + b2[e]; scale by routing weight; atomic add ----
  for (int nt = 0; nt < 8; ++nt) {
    int n = wave * 128 + nt * 16 + ln;
    float bias = b2[e * DM + n];
    v8f acc0, acc1;
#pragma unroll
    for (int r = 0; r < 8; ++r) { acc0[r] = bias; acc1[r] = bias; }
    const unsigned short* wrow = W2t + ((size_t)e * DM + n) * DF;
    if (nt < 7) __builtin_prefetch(wrow + 16 * DF, 0, 1);  // next N-tile row
#pragma unroll 2
    for (int k = 0; k < DF; k += 32) {
      AFrag bf;
      bf.q[0] = *(const uint4*)(wrow + k + bkoff);
      bf.q[1] = *(const uint4*)(wrow + k + bkoff + 16);
      AFrag a0, a1;
      int c0 = (k + akoff0) >> 3, c1 = (k + akoff1) >> 3;
      a0.q[0] = *(const uint4*)(Hs + ln * DF + ((c0 ^ ln) * 8));
      a0.q[1] = *(const uint4*)(Hs + ln * DF + ((c1 ^ ln) * 8));
      a1.q[0] = *(const uint4*)(Hs + (16 + ln) * DF + ((c0 ^ ln) * 8));
      a1.q[1] = *(const uint4*)(Hs + (16 + ln) * DF + ((c1 ^ ln) * 8));
      acc0 = __builtin_amdgcn_wmma_f32_16x16x32_bf16(false, a0.v, false, bf.v,
                                                     (short)0, acc0, false, false);
      acc1 = __builtin_amdgcn_wmma_f32_16x16x32_bf16(false, a1.v, false, bf.v,
                                                     (short)0, acc1, false, false);
    }
#pragma unroll
    for (int r = 0; r < 8; ++r) {
      int m0 = r + lh * 8;
      if (base + m0 < cnt) {
        int tok = tlist[m0];
        float w = combine[(size_t)tok * NE + e];
        unsafeAtomicAdd(out + (size_t)tok * DM + n, acc0[r] * w);
      }
      int m1 = m0 + 16;
      if (base + m1 < cnt) {
        int tok = tlist[m1];
        float w = combine[(size_t)tok * NE + e];
        unsafeAtomicAdd(out + (size_t)tok * DM + n, acc1[r] * w);
      }
    }
  }
}

// ---------------------------------------------------------------------------
// Host launcher
// ---------------------------------------------------------------------------
extern "C" void kernel_launch(void* const* d_in, const int* in_sizes, int n_in,
                              void* d_out, int out_size, void* d_ws, size_t ws_size,
                              hipStream_t stream) {
  (void)in_sizes; (void)n_in; (void)ws_size;
  const float* x  = (const float*)d_in[0];
  const float* W1 = (const float*)d_in[1];
  const float* b1 = (const float*)d_in[2];
  const float* W2 = (const float*)d_in[3];
  const float* b2 = (const float*)d_in[4];
  const float* Wg = (const float*)d_in[5];
  const float* bg = (const float*)d_in[6];
  float* out = (float*)d_out;

  // workspace layout (~151.5 MB)
  char* ws = (char*)d_ws;
  unsigned short* W1t  = (unsigned short*)(ws);                    // 64 MB
  unsigned short* W2t  = (unsigned short*)(ws + 67108864);         // 64 MB
  unsigned short* xb   = (unsigned short*)(ws + 134217728);        // 16 MB
  float*          comb = (float*)(ws + 150994944);                 // 256 KB
  int*            toks = (int*)(ws + 151257088);                   // 256 KB
  int*            cnts = (int*)(ws + 151519232);                   // 32 B

  (void)hipMemsetAsync(out, 0, (size_t)out_size * sizeof(float), stream);
  (void)hipMemsetAsync(cnts, 0, NE * sizeof(int), stream);

  // W1[e][d][f] -> W1t[e][f][d] ; W2[e][f][d] -> W2t[e][d][f] ; x -> bf16
  transpose_bf16<<<dim3(DF / 32, DM / 32, NE), dim3(32, 8), 0, stream>>>(W1, W1t, DM, DF);
  transpose_bf16<<<dim3(DM / 32, DF / 32, NE), dim3(32, 8), 0, stream>>>(W2, W2t, DF, DM);
  convert_bf16<<<(NT * DM) / (256 * 8), 256, 0, stream>>>(x, xb);

  router_kernel<<<NT / 8, 256, 0, stream>>>(x, Wg, bg, comb);
  assign_kernel<<<NT / 256, 256, 0, stream>>>(comb, cnts, toks);

  const size_t lds_bytes = (size_t)ROWS * DM * 2 + (size_t)ROWS * DF * 2;  // 320 KB
  (void)hipFuncSetAttribute((const void*)moe_expert_kernel,
                            hipFuncAttributeMaxDynamicSharedMemorySize, (int)lds_bytes);
  moe_expert_kernel<<<dim3(NT / ROWS, NE), 256, lds_bytes, stream>>>(
      xb, W1t, b1, W2t, b2, comb, cnts, toks, out);
}
